// SlidingWindowAttention_39015482917100
// MI455X (gfx1250) — 16-bit and 8-bit matrix operand order repaired and checked by reading, compiled, not run on hardware
//
#include <hip/hip_runtime.h>

constexpr int TQ  = 2048;
constexpr int CQ  = 1024;
constexpr int HQ  = 16;
constexpr int HDQ = 64;
constexpr int WINQ = 256;
constexpr float QSTEP = 2.0f / 255.0f;

typedef __bf16 bf16_t;
typedef __attribute__((ext_vector_type(16))) __bf16 v16bf;
typedef __attribute__((ext_vector_type(8)))  float  v8f;
typedef __attribute__((ext_vector_type(8)))  int    v8i;
typedef __attribute__((ext_vector_type(4)))  int    v4i;

#define USE_ASYNC_LDS __has_builtin(__builtin_amdgcn_global_load_async_to_lds_b128)

#if USE_ASYNC_LDS
typedef __attribute__((address_space(1))) v4i g_v4i;
typedef __attribute__((address_space(3))) v4i l_v4i;
#endif


__device__ __forceinline__ v16bf load_a16(const bf16_t* rowp, int hh) {
    const v4i* p = (const v4i*)rowp;
    v4i c0 = p[2 * hh];
    v4i c1 = p[2 * hh + 1];
    v8i t = {c0[0], c0[1], c0[2], c0[3], c1[0], c1[1], c1[2], c1[3]};
    return __builtin_bit_cast(v16bf, t);
}

__device__ __forceinline__ void copy32_g2l(const bf16_t* g, bf16_t* l) {
#if USE_ASYNC_LDS
    __builtin_amdgcn_global_load_async_to_lds_b128((g_v4i*)g, (l_v4i*)l, 0, 0);
    __builtin_amdgcn_global_load_async_to_lds_b128((g_v4i*)g, (l_v4i*)l, 16, 0);
#else
    *(v16bf*)l = *(const v16bf*)g;
#endif
}

__device__ __forceinline__ void async_wait0() {
#if USE_ASYNC_LDS
#if __has_builtin(__builtin_amdgcn_s_wait_asynccnt)
    __builtin_amdgcn_s_wait_asynccnt(0);
#else
    asm volatile("s_wait_asynccnt 0x0" ::: "memory");
#endif
#endif
}

__device__ __forceinline__ float gmax16(float v) {
    v = fmaxf(v, __shfl_xor(v, 1));
    v = fmaxf(v, __shfl_xor(v, 2));
    v = fmaxf(v, __shfl_xor(v, 4));
    v = fmaxf(v, __shfl_xor(v, 8));
    return v;
}
__device__ __forceinline__ float gsum16(float v) {
    v += __shfl_xor(v, 1);
    v += __shfl_xor(v, 2);
    v += __shfl_xor(v, 4);
    v += __shfl_xor(v, 8);
    return v;
}

__device__ __forceinline__ int iclamp(int x, int lo, int hi) {
    return x < lo ? lo : (x > hi ? hi : x);
}

__global__ void cvt_bf16_kernel(const float* __restrict__ src,
                                bf16_t* __restrict__ dst, int n) {
    int i = blockIdx.x * blockDim.x + threadIdx.x;
    if (i < n) dst[i] = (bf16_t)src[i];
}

__global__ void cvt_transpose_kernel(const float* __restrict__ src,
                                     bf16_t* __restrict__ dst,
                                     int rows, int cols) {
    int i = blockIdx.x * blockDim.x + threadIdx.x;
    if (i < rows * cols) {
        int r = i / cols, c = i % cols;
        dst[(size_t)c * rows + r] = (bf16_t)src[i];
    }
}

__device__ __forceinline__ void gemm_core_64x64(const bf16_t* __restrict__ aM,
                                                const bf16_t* __restrict__ bT,
                                                int m0, int n0, int K,
                                                bf16_t (*bsh)[64 * 32],
                                                v8f acc[4]) {
    const int t    = threadIdx.x;
    const int lane = t & 31;
    const int wave = t >> 5;
    const int mq = lane & 15, hh = lane >> 4;

    const int bn = t >> 1, bhalf = t & 1;
    const bf16_t* bsrc = bT + (size_t)(n0 + bn) * K + bhalf * 16;
    bf16_t* bdst[2] = {&bsh[0][bn * 32 + bhalf * 16],
                       &bsh[1][bn * 32 + bhalf * 16]};

    copy32_g2l(bsrc, bdst[0]);

    const bf16_t* arow = aM + (size_t)(m0 + wave * 16 + mq) * K;

    for (int k0 = 0; k0 < K; k0 += 32) {
        const int buf = (k0 >> 5) & 1;
        async_wait0();
        __syncthreads();
        if (k0 + 32 < K)
            copy32_g2l(bsrc + k0 + 32, bdst[buf ^ 1]);

        __builtin_prefetch(arow + k0 + 128, 0, 1);
        v16bf a = load_a16(arow + k0, hh);
        const bf16_t* bb = &bsh[buf][0];
        #pragma unroll
        for (int s = 0; s < 4; ++s) {
            v16bf b = *(const v16bf*)(bb + (s * 16 + mq) * 32 + hh * 16);
            acc[s] = __builtin_amdgcn_wmma_f32_16x16x32_bf16(
                false, a, false, b, (short)0, acc[s], false, false);
        }
    }
}

__global__ __launch_bounds__(128)
void gemm1_qkv_quant_kernel(const bf16_t* __restrict__ xb,
                            const bf16_t* __restrict__ waT,
                            signed char* __restrict__ q8,
                            signed char* __restrict__ k8,
                            bf16_t* __restrict__ vT) {
    __shared__ bf16_t bsh[2][64 * 32];

    const int lane = threadIdx.x & 31;
    const int wave = threadIdx.x >> 5;
    const int m0 = blockIdx.x * 64;
    const int n0 = blockIdx.y * 64;
    const int mq = lane & 15, hh = lane >> 4;

    v8f acc[4] = {v8f{}, v8f{}, v8f{}, v8f{}};
    gemm_core_64x64(xb, waT, m0, n0, CQ, bsh, acc);

    const int sel = n0 / CQ;
    const int h   = (n0 % CQ) / HDQ;
    #pragma unroll
    for (int s = 0; s < 4; ++s) {
        #pragma unroll
        for (int r = 0; r < 8; ++r) {
            int row = m0 + wave * 16 + r + hh * 8;
            int d   = s * 16 + mq;
            float f = acc[s][r];
            f = fminf(1.0f, fmaxf(-1.0f, f));
            int iq = iclamp((int)rintf(f * (1.0f / QSTEP)), -127, 127);
            if (sel == 0)
                q8[((size_t)h * TQ + row) * HDQ + d] = (signed char)iq;
            else if (sel == 1)
                k8[((size_t)h * TQ + row) * HDQ + d] = (signed char)iq;
            else
                vT[((size_t)h * HDQ + d) * TQ + row] = (bf16_t)(float)iq;
        }
    }
}

__global__ __launch_bounds__(32)
void swa_kernel(const signed char* __restrict__ q8,
                const signed char* __restrict__ k8,
                const bf16_t* __restrict__ vT,
                bf16_t* __restrict__ ya) {
    __shared__ bf16_t plds[16 * 32];

    const int lane = threadIdx.x;
    const int q0 = blockIdx.x * 16;
    const int h  = blockIdx.y;
    const int mq = lane & 15, hh = lane >> 4;

    const unsigned long long* qp =
        (const unsigned long long*)(q8 + ((size_t)h * TQ + q0 + mq) * HDQ);
    unsigned long long c0 = qp[2 * hh], c1 = qp[2 * hh + 1], c2 = qp[4 + 2 * hh], c3 = qp[5 + 2 * hh];
    v8i aq = {(int)c0, (int)(c0 >> 32), (int)c1, (int)(c1 >> 32),
              (int)c2, (int)(c2 >> 32), (int)c3, (int)(c3 >> 32)};

    v8f o[4] = {v8f{}, v8f{}, v8f{}, v8f{}};
    float mrun[8], lrun[8];
    #pragma unroll
    for (int r = 0; r < 8; ++r) { mrun[r] = -1e30f; lrun[r] = 0.0f; }

    const float sscale = QSTEP * QSTEP * 0.125f;

    for (int p = 0; p < 9; ++p) {
        int kb = q0 - WINQ + p * 32;
        float sc[2][8];

        #pragma unroll
        for (int j = 0; j < 2; ++j) {
            int kt = kb + j * 16;
            int key = kt + mq;
            int keyc = iclamp(key, 0, TQ - 1);
            const v4i* kp = (const v4i*)(k8 + ((size_t)h * TQ + keyc) * HDQ);
            v4i b0 = kp[hh], b1 = kp[2 + hh];
            v8i bk = {b0[0], b0[1], b0[2], b0[3], b1[0], b1[1], b1[2], b1[3]};
            v8i cz = {};
            v8i si = __builtin_amdgcn_wmma_i32_16x16x64_iu8(
                true, aq, true, bk, cz, false, false);
            #pragma unroll
            for (int r = 0; r < 8; ++r) {
                int irow = q0 + r + hh * 8;
                float f = (float)si[r] * sscale;
                f = fminf(1.0f, fmaxf(-1.0f, f));
                f = QSTEP * rintf(f * (1.0f / QSTEP));
                bool ok = (key <= irow) && (key > irow - WINQ) && (key >= 0);
                sc[j][r] = ok ? f : -3e38f;
            }
        }

        __syncthreads();
        #pragma unroll
        for (int r = 0; r < 8; ++r) {
            float tm   = gmax16(fmaxf(sc[0][r], sc[1][r]));
            float mnew = fmaxf(mrun[r], tm);
            float ea = __expf(sc[0][r] - mnew);
            float eb = __expf(sc[1][r] - mnew);
            float corr = __expf(mrun[r] - mnew);
            lrun[r] = lrun[r] * corr + gsum16(ea + eb);
            mrun[r] = mnew;
            o[0][r] *= corr; o[1][r] *= corr; o[2][r] *= corr; o[3][r] *= corr;
            int row = r + hh * 8;
            plds[row * 32 + mq]      = (bf16_t)ea;
            plds[row * 32 + 16 + mq] = (bf16_t)eb;
        }
        __syncthreads();

        v16bf ap = load_a16(plds + mq * 32, hh);
        int kbase = iclamp(kb + hh * 16, 0, TQ - 16);
        #pragma unroll
        for (int t = 0; t < 4; ++t) {
            int d = t * 16 + mq;
            v16bf bv = *(const v16bf*)(vT + ((size_t)h * HDQ + d) * TQ + kbase);
            o[t] = __builtin_amdgcn_wmma_f32_16x16x32_bf16(
                false, ap, false, bv, (short)0, o[t], false, false);
        }
    }

    #pragma unroll
    for (int t = 0; t < 4; ++t) {
        #pragma unroll
        for (int r = 0; r < 8; ++r) {
            float y = o[t][r] * (QSTEP / lrun[r]);
            y = fminf(1.0f, fmaxf(-1.0f, y));
            y = QSTEP * rintf(y * (1.0f / QSTEP));
            int row = q0 + r + hh * 8;
            ya[(size_t)row * CQ + h * HDQ + t * 16 + mq] = (bf16_t)y;
        }
    }
}

__global__ __launch_bounds__(128)
void gemm2_kernel(const bf16_t* __restrict__ ya,
                  const bf16_t* __restrict__ wpT,
                  float* __restrict__ out) {
    __shared__ bf16_t bsh[2][64 * 32];

    const int lane = threadIdx.x & 31;
    const int wave = threadIdx.x >> 5;
    const int m0 = blockIdx.x * 64;
    const int n0 = blockIdx.y * 64;
    const int mq = lane & 15, hh = lane >> 4;

    v8f acc[4] = {v8f{}, v8f{}, v8f{}, v8f{}};
    gemm_core_64x64(ya, wpT, m0, n0, CQ, bsh, acc);

    #pragma unroll
    for (int s = 0; s < 4; ++s)
        #pragma unroll
        for (int r = 0; r < 8; ++r)
            out[(size_t)(m0 + wave * 16 + r + hh * 8) * CQ + n0 + s * 16 + mq] =
                acc[s][r];
}

extern "C" void kernel_launch(void* const* d_in, const int* in_sizes, int n_in,
                              void* d_out, int out_size, void* d_ws, size_t ws_size,
                              hipStream_t stream) {
    (void)in_sizes; (void)n_in; (void)out_size; (void)ws_size;
    const float* x      = (const float*)d_in[0];
    const float* w_attn = (const float*)d_in[1];
    const float* w_proj = (const float*)d_in[2];
    float* out = (float*)d_out;

    char* ws = (char*)d_ws;
    bf16_t*      xb  = (bf16_t*)(ws);
    bf16_t*      waT = (bf16_t*)(ws + (size_t)4  * 1024 * 1024);
    bf16_t*      wpT = (bf16_t*)(ws + (size_t)10 * 1024 * 1024);
    signed char* q8  = (signed char*)(ws + (size_t)12 * 1024 * 1024);
    signed char* k8  = (signed char*)(ws + (size_t)14 * 1024 * 1024);
    bf16_t*      vT  = (bf16_t*)(ws + (size_t)16 * 1024 * 1024);
    bf16_t*      ya  = (bf16_t*)(ws + (size_t)20 * 1024 * 1024);

    int nx = TQ * CQ;
    cvt_bf16_kernel<<<(nx + 255) / 256, 256, 0, stream>>>(x, xb, nx);

    int nwa = CQ * 3 * CQ;
    cvt_transpose_kernel<<<(nwa + 255) / 256, 256, 0, stream>>>(w_attn, waT, CQ, 3 * CQ);
    int nwp = CQ * CQ;
    cvt_transpose_kernel<<<(nwp + 255) / 256, 256, 0, stream>>>(w_proj, wpT, CQ, CQ);

    gemm1_qkv_quant_kernel<<<dim3(TQ / 64, 3 * CQ / 64), 128, 0, stream>>>(
        xb, waT, q8, k8, vT);

    swa_kernel<<<dim3(TQ / 16, HQ), 32, 0, stream>>>(q8, k8, vT, ya);

    gemm2_kernel<<<dim3(TQ / 64, CQ / 64), 128, 0, stream>>>(ya, wpT, out);
}
